// LshSpatialAttention_10093173145852
// MI455X (gfx1250) — matrix operand order repaired and checked by reading, not compiled, not run on hardware
//
#include <hip/hip_runtime.h>
#include <hip/hip_bf16.h>

typedef __attribute__((ext_vector_type(16))) __bf16 v16bf;
typedef __attribute__((ext_vector_type(8)))  float  v8f;
typedef int v4i_vs __attribute__((vector_size(16)));

#define BL     192      // b*l
#define NV     325      // valid tokens
#define NP     352      // padded tokens
#define DF     256      // d_feat
#define DPOS   64
#define DP     320      // d_feat + d_pos
#define NHEADS 8
#define EH     32       // head dim
#define NHASH  4
#define NBUK   22
#define NCH    88       // NHASH*NBUK chunks per bh row
#define BKT    16       // bucket size
#define TOT    1408     // NHASH*NP
#define BHT    1536     // BL*NHEADS
#define M1     (BL*NP)  // 67584

#define SA_STRIDE 328   // 16-row A tile stride (elements); 328*2B => 16 distinct banks

#if defined(__has_builtin)
#if __has_builtin(__builtin_amdgcn_global_load_async_to_lds_b128)
#define HAVE_ASYNC_LDS 1
#endif
#if __has_builtin(__builtin_amdgcn_s_wait_asynccnt)
#define HAVE_WAIT_ASYNC_BUILTIN 1
#endif
#endif

__device__ __forceinline__ float bf2f(unsigned short h) {
  unsigned u = ((unsigned)h) << 16;
  return __builtin_bit_cast(float, u);
}
__device__ __forceinline__ unsigned short f2bf(float f) {
  unsigned u = __builtin_bit_cast(unsigned, f);
  unsigned r = u + 0x7FFFu + ((u >> 16) & 1u);
  return (unsigned short)(r >> 16);
}

__device__ __forceinline__ v8f wmma_bf16(v16bf a, v16bf b, v8f c) {
  return __builtin_amdgcn_wmma_f32_16x16x32_bf16(false, a, false, b, (short)0, c,
                                                 false, false);
}

// A fragment: 16x32 bf16, row-major source with element stride `stride`.
// Layout: lanes 0-15 rows M=0..15; VGPR v: K=2v,2v+1;
// lanes 16-31 same rows, K=16+2v,16+2v+1.
__device__ __forceinline__ v16bf load_a_frag(const unsigned short* base, int stride, int lane) {
  union { v16bf v; unsigned u[8]; } r;
  int m = lane & 15, half = lane >> 4;
#pragma unroll
  for (int v = 0; v < 8; ++v) {
    int kb = 2 * v + 16 * half;
    r.u[v] = *(const unsigned*)(base + (size_t)m * stride + kb);
  }
  return r.v;
}
// B fragment: 32x16 bf16 matrix B[K][N], sourced from transposed storage BT[N][K]
// (row-major, element stride `stride`). VGPR v: lanes 0-15 col=lane, K=2v,2v+1;
// lanes 16-31 col=lane-16, K=16+2v,16+2v+1.
__device__ __forceinline__ v16bf load_b_frag(const unsigned short* bt, int stride, int lane) {
  union { v16bf v; unsigned u[8]; } r;
  int n = lane & 15, half = lane >> 4;
#pragma unroll
  for (int v = 0; v < 8; ++v) {
    int k2 = 2 * v + 16 * half;
    r.u[v] = *(const unsigned*)(bt + (size_t)n * stride + k2);
  }
  return r.v;
}

// ---------------------------------------------------------------------------
// bf16 WMMA GEMM: out = A[M,K] * BT[N,K]^T (+bias)
// Block = 128 threads (4 waves). Block stages the 16xK A tile in LDS once
// (async-to-LDS when available), each wave computes a 16x64 strip:
// 4 accumulators reuse one A fragment per K step (4 WMMA per A load).
// mode 0: plain store (f32 and/or bf16)
// mode 1: rows with (row % npad) >= nvalid forced to 0 (post-proj padding)
// mode 2: store f32 remapped to [bi, n<nvalid, col], skipping pad rows
// ---------------------------------------------------------------------------
__global__ void __launch_bounds__(128)
gemm_bf16_wmma(const unsigned short* __restrict__ A,
               const unsigned short* __restrict__ BT,
               const float* __restrict__ bias,
               int Mdim, int Ndim, int Kdim,
               float* __restrict__ outF,
               unsigned short* __restrict__ outB,
               int mode, int nvalid, int npad) {
  __shared__ unsigned short sA[16][SA_STRIDE];

  int tid  = threadIdx.x;
  int lane = tid & 31;
  int wid  = tid >> 5;
  int tm   = blockIdx.x;

  // ---- stage A tile (16 rows x Kdim) into LDS ----
  const unsigned short* Abase = A + (size_t)tm * 16 * Kdim;
  int cpr    = Kdim >> 3;          // 16B chunks per row
  int chunks = 16 * cpr;
#ifdef HAVE_ASYNC_LDS
  for (int ch = tid; ch < chunks; ch += 128) {
    int r = ch / cpr, c = ch % cpr;
    __builtin_amdgcn_global_load_async_to_lds_b128(
        (__attribute__((address_space(1))) v4i_vs*)(Abase + (size_t)r * Kdim + c * 8),
        (__attribute__((address_space(3))) v4i_vs*)(&sA[r][c * 8]),
        0, 0);
  }
#ifdef HAVE_WAIT_ASYNC_BUILTIN
  __builtin_amdgcn_s_wait_asynccnt(0);
#else
  asm volatile("s_wait_asynccnt 0" ::: "memory");
#endif
#else
  for (int ch = tid; ch < chunks; ch += 128) {
    int r = ch / cpr, c = ch % cpr;
    uint4 d = *(const uint4*)(Abase + (size_t)r * Kdim + c * 8);
    *(uint4*)(&sA[r][c * 8]) = d;
  }
#endif
  __syncthreads();

  int ntile0 = (blockIdx.y * 4 + wid) * 4;   // first 16-col tile of this wave
  if (ntile0 * 16 >= Ndim) return;

  int m = lane & 15, half = lane >> 4;
  v8f acc0 = {0.f,0.f,0.f,0.f,0.f,0.f,0.f,0.f};
  v8f acc1 = acc0, acc2 = acc0, acc3 = acc0;

  const unsigned short* br0 = BT + (size_t)((ntile0 + 0) * 16 + m) * Kdim;
  const unsigned short* br1 = BT + (size_t)((ntile0 + 1) * 16 + m) * Kdim;
  const unsigned short* br2 = BT + (size_t)((ntile0 + 2) * 16 + m) * Kdim;
  const unsigned short* br3 = BT + (size_t)((ntile0 + 3) * 16 + m) * Kdim;

  for (int k = 0; k < Kdim; k += 32) {
    v16bf a = load_a_frag(&sA[0][k], SA_STRIDE, lane);
    union { v16bf v; unsigned u[8]; } fb0, fb1, fb2, fb3;
#pragma unroll
    for (int v = 0; v < 8; ++v) {
      int k2 = 2 * v + 16 * half;
      fb0.u[v] = *(const unsigned*)(br0 + k + k2);
      fb1.u[v] = *(const unsigned*)(br1 + k + k2);
      fb2.u[v] = *(const unsigned*)(br2 + k + k2);
      fb3.u[v] = *(const unsigned*)(br3 + k + k2);
    }
    acc0 = wmma_bf16(a, fb0.v, acc0);
    acc1 = wmma_bf16(a, fb1.v, acc1);
    acc2 = wmma_bf16(a, fb2.v, acc2);
    acc3 = wmma_bf16(a, fb3.v, acc3);
  }

#pragma unroll
  for (int t = 0; t < 4; ++t) {
    v8f acc = (t == 0) ? acc0 : (t == 1) ? acc1 : (t == 2) ? acc2 : acc3;
#pragma unroll
    for (int r = 0; r < 8; ++r) {
      int gr = tm * 16 + r + 8 * half;      // C layout: VGPR r -> M = 8*(lane/16)+r
      int gc = (ntile0 + t) * 16 + m;       // N = lane%16
      float val = acc[r] + (bias ? bias[gc] : 0.0f);
      if (mode == 1) { int n = gr % npad; if (n >= nvalid) val = 0.0f; }
      if (mode == 2) {
        int n = gr % npad;
        if (n < nvalid) {
          int bi = gr / npad;
          outF[((size_t)bi * nvalid + n) * Ndim + gc] = val;
        }
      } else {
        size_t o = (size_t)gr * Ndim + gc;
        if (outF) outF[o] = val;
        if (outB) outB[o] = f2bf(val);
      }
    }
  }
}

// ---------------------------------------------------------------------------
// Weight transpose + fp32 -> bf16:  wT[n*K + k] = w[k*N + n]
// ---------------------------------------------------------------------------
__global__ void convert_wT(const float* __restrict__ wsrc,
                           unsigned short* __restrict__ wT, int K, int N) {
  int idx = blockIdx.x * blockDim.x + threadIdx.x;
  if (idx >= K * N) return;
  int k = idx / N, n = idx % N;
  wT[(size_t)n * K + k] = f2bf(wsrc[idx]);
}

// ---------------------------------------------------------------------------
// Build A1 = concat(x, ste) with zero pad rows, bf16
// ---------------------------------------------------------------------------
__global__ void prep_concat(const float* __restrict__ x,
                            const float* __restrict__ ste,
                            unsigned short* __restrict__ A1) {
  size_t idx = (size_t)blockIdx.x * blockDim.x + threadIdx.x;
  if (idx >= (size_t)M1 * DP) return;
  int c   = (int)(idx % DP);
  int row = (int)(idx / DP);
  int n = row % NP, bi = row / NP;
  float v = 0.f;
  if (n < NV) {
    if (c < DF) v = x[((size_t)bi * NV + n) * DF + c];
    else        v = ste[((size_t)bi * NV + n) * DPOS + (c - DF)];
  }
  A1[idx] = f2bf(v);
}

// ---------------------------------------------------------------------------
// LSH bucketing: rot = qk . rotations, argmax over [rot, -rot]
// ---------------------------------------------------------------------------
__global__ void buckets_kernel(const unsigned short* __restrict__ qk,
                               const float* __restrict__ rot,
                               unsigned char* __restrict__ bkt) {
  int idx = blockIdx.x * blockDim.x + threadIdx.x;
  if (idx >= BHT * NHASH * NP) return;
  int n  = idx % NP;
  int h  = (idx / NP) % NHASH;
  int bh = idx / (NP * NHASH);
  int bi = bh >> 3, head = bh & 7;
  const unsigned short* q = qk + ((size_t)(bi * NP + n) * DF + head * EH);
  float qv[EH];
#pragma unroll
  for (int e = 0; e < EH; ++e) qv[e] = bf2f(q[e]);
  float s[11];
#pragma unroll
  for (int r = 0; r < 11; ++r) {
    float acc = 0.f;
#pragma unroll
    for (int e = 0; e < EH; ++e)
      acc += qv[e] * rot[(size_t)e * (NHASH * 11) + h * 11 + r];
    s[r] = acc;
  }
  float best = s[0]; int bid = 0;
  for (int r = 1; r < 11; ++r) if (s[r] > best) { best = s[r]; bid = r; }
  for (int r = 0; r < 11; ++r) { float v = -s[r]; if (v > best) { best = v; bid = 11 + r; } }
  bkt[(size_t)bh * TOT + h * NP + n] = (unsigned char)(h * NBUK + bid);
}

// ---------------------------------------------------------------------------
// Stable counting sort per bh row (key = bucket, tie-break by ticker) -> sticker
// ---------------------------------------------------------------------------
__global__ void __launch_bounds__(128)
sort_kernel(const unsigned char* __restrict__ buckets,
            unsigned short* __restrict__ st) {
  int bh = blockIdx.x;
  __shared__ unsigned cnt[NCH];
  __shared__ unsigned off[NCH];
  const unsigned char* bc = buckets + (size_t)bh * TOT;
  for (int i = threadIdx.x; i < NCH; i += blockDim.x) cnt[i] = 0;
  __syncthreads();
  for (int t = threadIdx.x; t < TOT; t += blockDim.x)
    atomicAdd(&cnt[bc[t]], 1u);
  __syncthreads();
  if (threadIdx.x == 0) {
    unsigned acc = 0;
    for (int i = 0; i < NCH; ++i) { off[i] = acc; acc += cnt[i]; }
  }
  __syncthreads();
  int b = threadIdx.x;
  if (b < NCH) {
    unsigned pos = off[b];
    unsigned short* out = st + (size_t)bh * TOT;
    for (int t = 0; t < TOT; ++t)
      if (bc[t] == (unsigned char)b) out[pos++] = (unsigned short)t;
  }
}

// ---------------------------------------------------------------------------
// Bucketed attention: 1 wave = 1 chunk (16 queries x 32 keys), WMMA dots + pV
// ---------------------------------------------------------------------------
__global__ void __launch_bounds__(128)
lsh_attention_kernel(const unsigned short* __restrict__ qk,
                     const unsigned short* __restrict__ vv,
                     const unsigned short* __restrict__ st,
                     float* __restrict__ logits,
                     unsigned short* __restrict__ obuf) {
  __shared__ unsigned short s_bq [4][16][32];
  __shared__ unsigned short s_bk [4][32][32];
  __shared__ unsigned short s_bvT[4][32][32];
  __shared__ unsigned short s_p  [4][16][32];
  __shared__ unsigned short s_nq [4][16];
  __shared__ unsigned short s_nk [4][32];
  __shared__ unsigned short s_tq [4][16];

  int w    = threadIdx.x >> 5;
  int lane = threadIdx.x & 31;
  int cid = blockIdx.x * 4 + w;
  int bh = cid / NCH;
  int c  = cid % NCH;
  int cprev = (c + NCH - 1) % NCH;
  int bi = bh >> 3, head = bh & 7;

  // lane j: key row j (j<16: current chunk; j>=16: previous chunk)
  int j  = lane;
  int cc = (j < 16) ? c : cprev;
  int t  = st[(size_t)bh * TOT + cc * 16 + (j & 15)];
  int n  = t % NP;
  s_nk[w][j] = (unsigned short)n;
  if (j < 16) { s_tq[w][j] = (unsigned short)t; s_nq[w][j] = (unsigned short)n; }

  const unsigned short* qrow = qk + ((size_t)(bi * NP + n) * DF + head * EH);
  const unsigned short* vrow = vv + ((size_t)(bi * NP + n) * DF + head * EH);
  float vals[EH]; float ss = 0.f;
#pragma unroll
  for (int e = 0; e < EH; ++e) { vals[e] = bf2f(qrow[e]); ss += vals[e] * vals[e]; }
  float inv = 1.0f / fmaxf(sqrtf(ss), 1e-12f);
#pragma unroll
  for (int e = 0; e < EH; ++e) {
    s_bk[w][j][e]  = f2bf(vals[e] * inv);   // normalized keys
    s_bvT[w][e][j] = vrow[e];               // values, transposed
    if (j < 16) s_bq[w][j][e] = qrow[e];    // raw queries
  }
  __syncthreads();

  v8f z = {0.f, 0.f, 0.f, 0.f, 0.f, 0.f, 0.f, 0.f};
  v16bf a  = load_a_frag(&s_bq[w][0][0], 32, lane);
  v16bf b0 = load_b_frag(&s_bk[w][0][0],  32, lane);   // keys 0..15
  v16bf b1 = load_b_frag(&s_bk[w][16][0], 32, lane);   // keys 16..31
  v8f c0 = wmma_bf16(a, b0, z);
  v8f c1 = wmma_bf16(a, b1, z);

  const float SCALE = 0.17677669529663687f;  // 32^-0.5
  int colN = lane & 15, half = lane >> 4;
#pragma unroll
  for (int r = 0; r < 8; ++r) {
    int row = r + 8 * half;
    float d0 = c0[r] * SCALE;
    if (s_nq[w][row] == s_nk[w][colN])      d0 = -50000.0f;
    float d1 = c1[r] * SCALE;
    if (s_nq[w][row] == s_nk[w][16 + colN]) d1 = -50000.0f;
    float m = fmaxf(d0, d1);
    for (int off = 1; off < 16; off <<= 1) m = fmaxf(m, __shfl_xor(m, off, 32));
    float e0 = __expf(d0 - m), e1 = __expf(d1 - m);
    float s = e0 + e1;
    for (int off = 1; off < 16; off <<= 1) s += __shfl_xor(s, off, 32);
    float invs = 1.0f / s;
    s_p[w][row][colN]      = f2bf(e0 * invs);
    s_p[w][row][16 + colN] = f2bf(e1 * invs);
    if (colN == 0)
      logits[(size_t)bh * TOT + s_tq[w][row]] = m + __logf(s);
  }
  __syncthreads();

  v16bf pa = load_a_frag(&s_p[w][0][0], 32, lane);
  v16bf w0 = load_b_frag(&s_bvT[w][0][0],  32, lane);  // e cols 0..15
  v16bf w1 = load_b_frag(&s_bvT[w][16][0], 32, lane);  // e cols 16..31
  v8f o0 = wmma_bf16(pa, w0, z);
  v8f o1 = wmma_bf16(pa, w1, z);
#pragma unroll
  for (int r = 0; r < 8; ++r) {
    int row = r + 8 * half;
    unsigned short* op = obuf + ((size_t)bh * TOT + s_tq[w][row]) * EH;
    op[colN]      = f2bf(o0[r]);   // scatter == undo-argsort
    op[16 + colN] = f2bf(o1[r]);
  }
}

// ---------------------------------------------------------------------------
// Combine the 4 hash rounds with softmax(logits) weights
// ---------------------------------------------------------------------------
__global__ void __launch_bounds__(128)
combine_kernel(const float* __restrict__ logits,
               const unsigned short* __restrict__ obuf,
               unsigned short* __restrict__ att) {
  int idx  = blockIdx.x * 4 + (threadIdx.x >> 5);
  int lane = threadIdx.x & 31;
  int bh = idx / NP, n = idx % NP;
  float l0 = logits[(size_t)bh * TOT + 0 * NP + n];
  float l1 = logits[(size_t)bh * TOT + 1 * NP + n];
  float l2 = logits[(size_t)bh * TOT + 2 * NP + n];
  float l3 = logits[(size_t)bh * TOT + 3 * NP + n];
  float m = fmaxf(fmaxf(l0, l1), fmaxf(l2, l3));
  float p0 = __expf(l0 - m), p1 = __expf(l1 - m);
  float p2 = __expf(l2 - m), p3 = __expf(l3 - m);
  float s = p0 + p1 + p2 + p3;
  float acc = p0 * bf2f(obuf[((size_t)bh * TOT + 0 * NP + n) * EH + lane])
            + p1 * bf2f(obuf[((size_t)bh * TOT + 1 * NP + n) * EH + lane])
            + p2 * bf2f(obuf[((size_t)bh * TOT + 2 * NP + n) * EH + lane])
            + p3 * bf2f(obuf[((size_t)bh * TOT + 3 * NP + n) * EH + lane]);
  acc /= s;
  int bi = bh >> 3, head = bh & 7;
  att[((size_t)(bi * NP + n)) * DF + head * EH + lane] = f2bf(acc);
}

// ---------------------------------------------------------------------------
extern "C" void kernel_launch(void* const* d_in, const int* in_sizes, int n_in,
                              void* d_out, int out_size, void* d_ws, size_t ws_size,
                              hipStream_t stream) {
  const float* x      = (const float*)d_in[0];
  const float* ste    = (const float*)d_in[1];
  const float* w_proj = (const float*)d_in[2];
  const float* b_proj = (const float*)d_in[3];
  const float* w_qk   = (const float*)d_in[4];
  const float* w_v    = (const float*)d_in[5];
  const float* w_out  = (const float*)d_in[6];
  const float* b_out  = (const float*)d_in[7];
  const float* rots   = (const float*)d_in[8];
  (void)in_sizes; (void)n_in; (void)out_size; (void)ws_size;

  char* p = (char*)d_ws;
  auto take = [&](size_t bytes) {
    char* r = p;
    p += (bytes + 255) & ~(size_t)255;
    return r;
  };
  unsigned short* A1     = (unsigned short*)take((size_t)M1 * DP * 2);
  unsigned short* h_bf   = (unsigned short*)take((size_t)M1 * DF * 2);
  unsigned short* qk_bf  = (unsigned short*)take((size_t)M1 * DF * 2);
  unsigned short* v_bf   = (unsigned short*)take((size_t)M1 * DF * 2);
  unsigned short* att    = (unsigned short*)take((size_t)M1 * DF * 2);
  unsigned short* wprojT = (unsigned short*)take((size_t)DF * DP * 2);
  unsigned short* wqkT   = (unsigned short*)take((size_t)DF * DF * 2);
  unsigned short* wvT    = (unsigned short*)take((size_t)DF * DF * 2);
  unsigned short* woutT  = (unsigned short*)take((size_t)DF * DF * 2);
  unsigned char*  bkt    = (unsigned char*) take((size_t)BHT * TOT);
  unsigned short* stbuf  = (unsigned short*)take((size_t)BHT * TOT * 2);
  float*          logits = (float*)         take((size_t)BHT * TOT * 4);
  unsigned short* obuf   = (unsigned short*)take((size_t)BHT * TOT * EH * 2);

  // 1) weights -> bf16 transposed
  convert_wT<<<(DP * DF + 255) / 256, 256, 0, stream>>>(w_proj, wprojT, DP, DF);
  convert_wT<<<(DF * DF + 255) / 256, 256, 0, stream>>>(w_qk, wqkT, DF, DF);
  convert_wT<<<(DF * DF + 255) / 256, 256, 0, stream>>>(w_v, wvT, DF, DF);
  convert_wT<<<(DF * DF + 255) / 256, 256, 0, stream>>>(w_out, woutT, DF, DF);

  // 2) concat(x, ste) with zero padding, bf16
  {
    size_t tot = (size_t)M1 * DP;
    prep_concat<<<(unsigned)((tot + 255) / 256), 256, 0, stream>>>(x, ste, A1);
  }

  // 3) h = A1 @ w_proj + b_proj (pad rows -> 0); block covers 256 cols
  gemm_bf16_wmma<<<dim3(M1 / 16, DF / 256), 128, 0, stream>>>(
      A1, wprojT, b_proj, M1, DF, DP, nullptr, h_bf, 1, NV, NP);

  // 4) qk, v
  gemm_bf16_wmma<<<dim3(M1 / 16, DF / 256), 128, 0, stream>>>(
      h_bf, wqkT, nullptr, M1, DF, DF, nullptr, qk_bf, 0, NV, NP);
  gemm_bf16_wmma<<<dim3(M1 / 16, DF / 256), 128, 0, stream>>>(
      h_bf, wvT, nullptr, M1, DF, DF, nullptr, v_bf, 0, NV, NP);

  // 5) LSH buckets
  buckets_kernel<<<(BHT * NHASH * NP + 255) / 256, 256, 0, stream>>>(qk_bf, rots, bkt);

  // 6) stable counting sort per bh row -> sticker
  sort_kernel<<<BHT, 128, 0, stream>>>(bkt, stbuf);

  // 7) chunked attention (WMMA), scattered (undone) output + per-position lse
  lsh_attention_kernel<<<(BHT * NCH) / 4, 128, 0, stream>>>(qk_bf, v_bf, stbuf,
                                                            logits, obuf);

  // 8) combine hash rounds
  combine_kernel<<<(BHT * NP) / 4, 128, 0, stream>>>(logits, obuf, att);

  // 9) out = att @ w_out + b_out, stored to d_out skipping pad rows
  gemm_bf16_wmma<<<dim3(M1 / 16, DF / 256), 128, 0, stream>>>(
      att, woutT, b_out, M1, DF, DF, (float*)d_out, nullptr, 2, NV, NP);
}
